// MultiHeadAttention_74792560492965
// MI455X (gfx1250) — matrix operand order repaired and checked by reading, not compiled, not run on hardware
//
#include <hip/hip_runtime.h>
#include <hip/hip_bf16.h>

#define DI __device__ __forceinline__

typedef _Float16 h16v __attribute__((ext_vector_type(16)));
typedef _Float16 h8v  __attribute__((ext_vector_type(8)));
typedef _Float16 h4v  __attribute__((ext_vector_type(4)));
typedef float    f8v  __attribute__((ext_vector_type(8)));
typedef float    f4v  __attribute__((ext_vector_type(4)));

constexpr int NB = 256;   // batch
constexpr int NT = 256;   // seq len
constexpr int NC = 384;   // embed
constexpr int NH = 6;     // heads
constexpr int HS = 64;    // head size
constexpr int QSZ = NB * NT * NC;  // elements per q/k/v buffer

// ---- CDNA5 async global->LDS copy (probe via __has_builtin, safe fallback) ----
#if defined(__has_builtin)
# if __has_builtin(__builtin_amdgcn_global_load_async_to_lds_b128)
#  define HAVE_ASYNC_B128 1
# endif
# if __has_builtin(__builtin_amdgcn_s_wait_asynccnt)
#  define HAVE_WAIT_ASYNC 1
# endif
#endif
#ifndef HAVE_ASYNC_B128
# define HAVE_ASYNC_B128 0
#endif
#ifndef HAVE_WAIT_ASYNC
# define HAVE_WAIT_ASYNC 0
#endif

// Builtin signature (from compiler diagnostic): param 1 is int4 in AS(1).
typedef int b128i __attribute__((vector_size(16)));
typedef __attribute__((address_space(1))) b128i* gb128p;
typedef __attribute__((address_space(3))) b128i* lb128p;

DI void g2l_b128(const _Float16* g, _Float16* l) {
#if HAVE_ASYNC_B128
  __builtin_amdgcn_global_load_async_to_lds_b128(
      (gb128p)(b128i*)(_Float16*)g, (lb128p)(b128i*)l, 0, 0);
#else
  *(h8v*)l = *(const h8v*)g;
#endif
}
DI void wait_async() {
#if HAVE_ASYNC_B128
# if HAVE_WAIT_ASYNC
  __builtin_amdgcn_s_wait_asynccnt(0);
# else
  asm volatile("s_wait_asynccnt 0" ::: "memory");
# endif
#endif
}

// ---------------- WMMA fragment helpers (CDNA5 16x16x32 f16 layouts) ----------
DI h16v frag_a(const _Float16* base, int stride, int row0, int k0, int lane) {
  const int lr = lane & 15, hi = lane >> 4;
  const _Float16* p = base + (row0 + lr) * stride + k0 + hi * 8;
  h8v lo = *(const h8v*)p;
  h8v hv = *(const h8v*)(p + 16);
  return __builtin_shufflevector(lo, hv, 0,1,2,3,4,5,6,7,8,9,10,11,12,13,14,15);
}
DI h16v frag_b(const _Float16* base, int stride, int n0, int k0, int lane) {
  const int lr = lane & 15, hi = lane >> 4;
  const _Float16* p = base + (n0 + lr) * stride + k0 + hi * 8;
  h8v lo = *(const h8v*)p;
  h8v hv = *(const h8v*)(p + 16);
  return __builtin_shufflevector(lo, hv, 0,1,2,3,4,5,6,7,8,9,10,11,12,13,14,15);
}
DI f8v wmma(f8v c, h16v a, h16v b) {
  return __builtin_amdgcn_wmma_f32_16x16x32_f16(false, a, false, b, (short)0, c,
                                                false, false);
}
DI h4v cvt4(f4v v) {
  h4v r; r.x = (_Float16)v.x; r.y = (_Float16)v.y;
  r.z = (_Float16)v.z; r.w = (_Float16)v.w; return r;
}

// ---------------- Kernel 1: fused QKV projection -------------------------------
// Block tile 128x128, 8 waves as 2x4, 64x32 per wave.  q scaled by 1/8.
// q,k stored [B,H,T,hs]; V stored TRANSPOSED [B,H,hs,T] (LDS-staged, coalesced).
__global__ __launch_bounds__(256)
void k_qkv(const float* __restrict__ x,
           const float* __restrict__ Wq, const float* __restrict__ Wk,
           const float* __restrict__ Wv,
           _Float16* __restrict__ qb, _Float16* __restrict__ kb,
           _Float16* __restrict__ vb) {
  __shared__ _Float16 As[128 * 40];
  __shared__ _Float16 Bs[128 * 40];
  __shared__ _Float16 Ts[128 * 136];  // V^T staging (cols x rows), pad to 136

  const int z = blockIdx.z;
  const float* W = (z == 0) ? Wq : (z == 1) ? Wk : Wv;
  const float scale = (z == 0) ? 0.125f : 1.0f;  // fold 1/sqrt(64) into q

  const int m0 = blockIdx.x * 128, n0 = blockIdx.y * 128;
  const int tid = threadIdx.x, lane = tid & 31, wave = tid >> 5;
  const int wm = wave & 1, wn = wave >> 1;   // 2x4 wave grid, 64x32 per wave
  const int lr = lane & 15, hi = lane >> 4;

  f8v acc[4][2] = {};
  for (int kk = 0; kk < NC; kk += 32) {
    __syncthreads();
    #pragma unroll
    for (int i = 0; i < 4; i++) {                    // x tile 128x32 f32
      int u = i * 256 + tid;
      int row = u >> 3, s = (u & 7) * 4;
      *(h4v*)(As + row * 40 + s) =
          cvt4(*(const f4v*)(x + (size_t)(m0 + row) * NC + kk + s));
    }
    #pragma unroll
    for (int i = 0; i < 4; i++) {                    // W tile 128x32 f32
      int u = i * 256 + tid;
      int row = u >> 3, s = (u & 7) * 4;
      *(h4v*)(Bs + row * 40 + s) =
          cvt4(*(const f4v*)(W + (size_t)(n0 + row) * NC + kk + s));
    }
    if (kk + 32 < NC) {                              // prefetch next k-tile
      __builtin_prefetch(x + (size_t)(m0 + (tid >> 3)) * NC + kk + 32, 0, 3);
      __builtin_prefetch(W + (size_t)(n0 + (tid >> 3)) * NC + kk + 32, 0, 3);
    }
    __syncthreads();
    h16v a[4], b[2];
    #pragma unroll
    for (int rf = 0; rf < 4; rf++) a[rf] = frag_a(As, 40, wm * 64 + rf * 16, 0, lane);
    #pragma unroll
    for (int cf = 0; cf < 2; cf++) b[cf] = frag_b(Bs, 40, wn * 32 + cf * 16, 0, lane);
    #pragma unroll
    for (int rf = 0; rf < 4; rf++)
      #pragma unroll
      for (int cf = 0; cf < 2; cf++)
        acc[rf][cf] = wmma(acc[rf][cf], a[rf], b[cf]);
  }

  if (z != 2) {
    _Float16* dst = (z == 0) ? qb : kb;
    #pragma unroll
    for (int rf = 0; rf < 4; rf++)
      #pragma unroll
      for (int cf = 0; cf < 2; cf++)
        #pragma unroll
        for (int r = 0; r < 8; r++) {
          int m = m0 + wm * 64 + rf * 16 + r + hi * 8;
          int col = n0 + wn * 32 + cf * 16 + lr;
          int b_ = m >> 8, t = m & 255;
          int h = col >> 6, d = col & 63;
          dst[(((size_t)b_ * NH + h) * NT + t) * HS + d] =
              (_Float16)(acc[rf][cf][r] * scale);
        }
  } else {
    // V: transpose 128x128 tile through LDS, store coalesced to [B,H,hs,T]
    __syncthreads();
    #pragma unroll
    for (int rf = 0; rf < 4; rf++)
      #pragma unroll
      for (int cf = 0; cf < 2; cf++)
        #pragma unroll
        for (int r = 0; r < 8; r++) {
          int rowl = wm * 64 + rf * 16 + r + hi * 8;     // t-local
          int coll = wn * 32 + cf * 16 + lr;             // d-local
          Ts[coll * 136 + rowl] = (_Float16)acc[rf][cf][r];
        }
    __syncthreads();
    const int b_ = m0 >> 8, t0 = m0 & 255;               // 128-row tile in one b
    #pragma unroll
    for (int i = 0; i < 8; i++) {                        // 128 d-rows x 16 chunks
      int u = i * 256 + tid;
      int dl = u >> 4, s = (u & 15) * 8;
      int col = n0 + dl, h = col >> 6, d = col & 63;
      *(h8v*)(vb + (((size_t)b_ * NH + h) * HS + d) * NT + t0 + s) =
          *(const h8v*)(Ts + dl * 136 + s);
    }
  }
}

// ---------------- Kernel 2: causal attention per (b,h), 64-row block -----------
// Dynamic LDS carve (122112 B):
//   S (64x264 f32) | P (64x264 f16) | Q (64x72 f16) | KV (64x72 f16) |
//   pmax (64x4) | psum (64x4) | rowsum (64)
constexpr int SMEM_ATTN =
    64 * 264 * 4 + 64 * 264 * 2 + 64 * 72 * 2 + 64 * 72 * 2 + 256 * 4 + 256 * 4 + 64 * 4;

__global__ __launch_bounds__(256)
void k_attn(_Float16* __restrict__ qbuf, const _Float16* __restrict__ kbuf,
            const _Float16* __restrict__ vbuf) {
  extern __shared__ char smem[];
  float*    S  = (float*)smem;                       // 67584 B
  _Float16* P  = (_Float16*)(smem + 67584);          // 33792 B
  _Float16* Q  = (_Float16*)(smem + 67584 + 33792);  //  9216 B
  _Float16* KV = Q + 64 * 72;                        //  9216 B
  float* pmax   = (float*)(KV + 64 * 72);
  float* psum   = pmax + 256;
  float* rowsum = psum + 256;

  const int bid = blockIdx.x;
  const int rb = bid & 3, bh = bid >> 2;
  const int r0 = rb * 64, nkb = rb + 1;              // causal: k-blocks 0..rb
  const size_t base = (size_t)bh * NT * HS;          // same base for [t][d] and [d][t]
  const int tid = threadIdx.x, lane = tid & 31, wave = tid >> 5;
  const int wr = wave & 3, wc = wave >> 2;
  const int lr = lane & 15, hi = lane >> 4;

  // Q tile (async copy: f16 in ws, no conversion needed)
  #pragma unroll
  for (int i = 0; i < 2; i++) {
    int u = i * 256 + tid;
    int row = u >> 3, s = (u & 7) * 8;
    g2l_b128(qbuf + base + (size_t)(r0 + row) * HS + s, Q + row * 72 + s);
  }

  // Phase 1: S = Q K^T for k-blocks 0..rb, exact causal mask on diagonal block
  for (int kbI = 0; kbI < nkb; kbI++) {
    __syncthreads();
    #pragma unroll
    for (int i = 0; i < 2; i++) {
      int u = i * 256 + tid;
      int row = u >> 3, s = (u & 7) * 8;
      g2l_b128(kbuf + base + (size_t)(kbI * 64 + row) * HS + s, KV + row * 72 + s);
    }
    wait_async();
    __syncthreads();
    f8v acc[2] = {};
    #pragma unroll
    for (int dstep = 0; dstep < 2; dstep++) {
      h16v a  = frag_a(Q, 72, wr * 16, dstep * 32, lane);
      h16v b0 = frag_b(KV, 72, wc * 32, dstep * 32, lane);
      h16v b1 = frag_b(KV, 72, wc * 32 + 16, dstep * 32, lane);
      acc[0] = wmma(acc[0], a, b0);
      acc[1] = wmma(acc[1], a, b1);
    }
    #pragma unroll
    for (int cf = 0; cf < 2; cf++)
      #pragma unroll
      for (int r = 0; r < 8; r++) {
        int row = wr * 16 + r + hi * 8;
        int coll = wc * 32 + cf * 16 + lr;
        float v = acc[cf][r];
        if (kbI * 64 + coll > r0 + row) v = -3.0e38f;   // causal mask
        S[row * 264 + kbI * 64 + coll] = v;
      }
  }
  __syncthreads();

  // Phase 2: one-pass softmax; unnormalized exp -> P (f16), sums kept f32
  const int Tc = nkb * 64, L = Tc >> 2;               // L in {16,32,48,64}
  const int row = tid >> 2, qq = tid & 3;
  const float* Srow = S + row * 264 + qq * L;
  _Float16*    Prow = P + row * 264 + qq * L;
  {
    float m = -3.4e38f;
    for (int j = 0; j < L; j += 4) {                  // vectorized ds_load_b128
      f4v v = *(const f4v*)(Srow + j);
      m = fmaxf(m, fmaxf(fmaxf(v.x, v.y), fmaxf(v.z, v.w)));
    }
    pmax[row * 4 + qq] = m;
  }
  __syncthreads();
  const float rmax = fmaxf(fmaxf(pmax[row * 4], pmax[row * 4 + 1]),
                           fmaxf(pmax[row * 4 + 2], pmax[row * 4 + 3]));
  {
    float s = 0.f;
    for (int j = 0; j < L; j += 4) {
      f4v v = *(const f4v*)(Srow + j);
      float e0 = __expf(v.x - rmax), e1 = __expf(v.y - rmax);
      float e2 = __expf(v.z - rmax), e3 = __expf(v.w - rmax);
      h4v p; p.x = (_Float16)e0; p.y = (_Float16)e1;
      p.z = (_Float16)e2; p.w = (_Float16)e3;
      *(h4v*)(Prow + j) = p;
      s += (e0 + e1) + (e2 + e3);
    }
    psum[row * 4 + qq] = s;
  }
  __syncthreads();
  if (qq == 0)
    rowsum[row] = psum[row * 4] + psum[row * 4 + 1] + psum[row * 4 + 2] + psum[row * 4 + 3];

  // Phase 3: O = P V.  V is pre-transposed [B,H,hs,T] -> straight tile copy.
  f8v oacc[2] = {};
  for (int kbI = 0; kbI < nkb; kbI++) {
    __syncthreads();
    #pragma unroll
    for (int i = 0; i < 2; i++) {
      int u = i * 256 + tid;
      int dl = u >> 3, s = (u & 7) * 8;
      g2l_b128(vbuf + base + (size_t)dl * NT + kbI * 64 + s, KV + dl * 72 + s);
    }
    wait_async();
    __syncthreads();
    #pragma unroll
    for (int step = 0; step < 2; step++) {
      h16v a  = frag_a(P, 264, wr * 16, kbI * 64 + step * 32, lane);
      h16v b0 = frag_b(KV, 72, wc * 32, step * 32, lane);
      h16v b1 = frag_b(KV, 72, wc * 32 + 16, step * 32, lane);
      oacc[0] = wmma(oacc[0], a, b0);
      oacc[1] = wmma(oacc[1], a, b1);
    }
  }
  // Epilogue: normalize, write f16 over the q buffer in place (only we read it)
  #pragma unroll
  for (int cf = 0; cf < 2; cf++)
    #pragma unroll
    for (int r = 0; r < 8; r++) {
      int lrow = wr * 16 + r + hi * 8;
      int d = wc * 32 + cf * 16 + lr;
      float v = oacc[cf][r] / rowsum[lrow];
      qbuf[base + (size_t)(r0 + lrow) * HS + d] = (_Float16)v;
    }
}

// ---------------- Kernel 3: output projection + bias ---------------------------
__global__ __launch_bounds__(256)
void k_proj(const _Float16* __restrict__ att, const float* __restrict__ Wp,
            const float* __restrict__ bp, float* __restrict__ out) {
  __shared__ _Float16 As[128 * 40];
  __shared__ _Float16 Bs[128 * 40];
  const int m0 = blockIdx.x * 128, n0 = blockIdx.y * 128;
  const int tid = threadIdx.x, lane = tid & 31, wave = tid >> 5;
  const int wm = wave & 1, wn = wave >> 1;
  const int lr = lane & 15, hi = lane >> 4;

  f8v acc[4][2] = {};
  for (int kk = 0; kk < NC; kk += 32) {
    __syncthreads();
    #pragma unroll
    for (int i = 0; i < 2; i++) {            // gather [B,H,T,hs] -> [m][c] tile
      int u = i * 256 + tid;
      int row = u >> 2, s = (u & 3) * 8;
      int m = m0 + row, b_ = m >> 8, t = m & 255;
      int h = kk >> 6, d0 = (kk & 63) + s;   // 32-wide slice stays inside a head
      *(h8v*)(As + row * 40 + s) =
          *(const h8v*)(att + (((size_t)b_ * NH + h) * NT + t) * HS + d0);
    }
    #pragma unroll
    for (int i = 0; i < 4; i++) {
      int u = i * 256 + tid;
      int rowN = u >> 3, s = (u & 7) * 4;
      *(h4v*)(Bs + rowN * 40 + s) =
          cvt4(*(const f4v*)(Wp + (size_t)(n0 + rowN) * NC + kk + s));
    }
    if (kk + 32 < NC)
      __builtin_prefetch(Wp + (size_t)(n0 + (tid >> 3)) * NC + kk + 32, 0, 3);
    __syncthreads();
    h16v a[4], b[2];
    #pragma unroll
    for (int rf = 0; rf < 4; rf++) a[rf] = frag_a(As, 40, wm * 64 + rf * 16, 0, lane);
    #pragma unroll
    for (int cf = 0; cf < 2; cf++) b[cf] = frag_b(Bs, 40, wn * 32 + cf * 16, 0, lane);
    #pragma unroll
    for (int rf = 0; rf < 4; rf++)
      #pragma unroll
      for (int cf = 0; cf < 2; cf++)
        acc[rf][cf] = wmma(acc[rf][cf], a[rf], b[cf]);
  }
  #pragma unroll
  for (int rf = 0; rf < 4; rf++)
    #pragma unroll
    for (int cf = 0; cf < 2; cf++) {
      int col = n0 + wn * 32 + cf * 16 + lr;
      float bias = bp[col];
      #pragma unroll
      for (int r = 0; r < 8; r++) {
        int m = m0 + wm * 64 + rf * 16 + r + hi * 8;
        out[(size_t)m * NC + col] = acc[rf][cf][r] + bias;
      }
    }
}

extern "C" void kernel_launch(void* const* d_in, const int* in_sizes, int n_in,
                              void* d_out, int out_size, void* d_ws, size_t ws_size,
                              hipStream_t stream) {
  const float* x  = (const float*)d_in[0];
  const float* Wk = (const float*)d_in[1];
  const float* Wq = (const float*)d_in[2];
  const float* Wv = (const float*)d_in[3];
  const float* Wp = (const float*)d_in[4];
  const float* bp = (const float*)d_in[5];
  float* out = (float*)d_out;

  _Float16* qb = (_Float16*)d_ws;          // 50.3 MB each, 151 MB total
  _Float16* kb = qb + QSZ;
  _Float16* vb = qb + 2 * (size_t)QSZ;

  dim3 blk(256);
  k_qkv<<<dim3(512, 3, 3), blk, 0, stream>>>(x, Wq, Wk, Wv, qb, kb, vb);
  k_attn<<<dim3(NB * NH * 4), blk, SMEM_ATTN, stream>>>(qb, kb, vb);
  k_proj<<<dim3(512, 3), blk, 0, stream>>>(qb, Wp, bp, out);
}
